// FP8Linear_68951404970065
// MI455X (gfx1250) — 8-bit matrix operand order repaired and checked by reading, compiled, not run on hardware
//
#include <hip/hip_runtime.h>

typedef __attribute__((ext_vector_type(16))) int       v16i;
typedef __attribute__((ext_vector_type(8)))  float     v8f;
typedef __attribute__((ext_vector_type(8)))  _Float16  v8h;
typedef __attribute__((ext_vector_type(4)))  unsigned int u32x4;
typedef __attribute__((ext_vector_type(2)))  unsigned int u32x2;

#define FP8_MAX 448.0f

__global__ void init_amax_kernel(unsigned int* amax) {
    if (threadIdx.x < 2) amax[threadIdx.x] = 0u;
}

__global__ __launch_bounds__(256) void amax_kernel(
    const _Float16* __restrict__ in, size_t n8, unsigned int* __restrict__ amax) {
    __shared__ float red[256];
    const v8h* p = (const v8h*)in;
    float m = 0.f;
    size_t stride = (size_t)gridDim.x * blockDim.x;
    for (size_t i = (size_t)blockIdx.x * blockDim.x + threadIdx.x; i < n8; i += stride) {
        v8h v = p[i];
#pragma unroll
        for (int j = 0; j < 8; ++j) m = fmaxf(m, fabsf((float)v[j]));
    }
    red[threadIdx.x] = m;
    __syncthreads();
    for (int s = 128; s > 0; s >>= 1) {
        if (threadIdx.x < s) red[threadIdx.x] = fmaxf(red[threadIdx.x], red[threadIdx.x + s]);
        __syncthreads();
    }
    if (threadIdx.x == 0) atomicMax(amax, __float_as_uint(red[0]));
}

__device__ inline unsigned int f32_to_e4m3_sw(float x) {
    unsigned int sign = (__float_as_uint(x) >> 24) & 0x80u;
    float ax = fminf(fabsf(x), FP8_MAX);
    if (ax == 0.f) return sign;
    int e;
    float m = frexpf(ax, &e);
    if (e <= -6) {
        return sign | __float2uint_rn(ax * 512.f);
    }
    unsigned int mant = __float2uint_rn((2.f * m - 1.f) * 8.f);
    unsigned int code = ((unsigned int)(e + 6) << 3) + mant;
    return sign | code;
}

__device__ inline unsigned int pack4_fp8(float f0, float f1, float f2, float f3) {
    f0 = fminf(fmaxf(f0, -FP8_MAX), FP8_MAX);
    f1 = fminf(fmaxf(f1, -FP8_MAX), FP8_MAX);
    f2 = fminf(fmaxf(f2, -FP8_MAX), FP8_MAX);
    f3 = fminf(fmaxf(f3, -FP8_MAX), FP8_MAX);
#if __has_builtin(__builtin_amdgcn_cvt_pk_fp8_f32)
    int v = 0;
    v = __builtin_amdgcn_cvt_pk_fp8_f32(f0, f1, v, false);
    v = __builtin_amdgcn_cvt_pk_fp8_f32(f2, f3, v, true);
    return (unsigned int)v;
#else
    return f32_to_e4m3_sw(f0) | (f32_to_e4m3_sw(f1) << 8) |
           (f32_to_e4m3_sw(f2) << 16) | (f32_to_e4m3_sw(f3) << 24);
#endif
}

__global__ __launch_bounds__(256) void quant_kernel(
    const _Float16* __restrict__ in, unsigned char* __restrict__ out,
    size_t n8, const unsigned int* __restrict__ amax_bits) {
    float amax  = fmaxf(__uint_as_float(*amax_bits), 1e-12f);
    float scale = FP8_MAX / amax;
    const v8h* p = (const v8h*)in;
    u32x2* q = (u32x2*)out;
    size_t stride = (size_t)gridDim.x * blockDim.x;
    for (size_t i = (size_t)blockIdx.x * blockDim.x + threadIdx.x; i < n8; i += stride) {
        v8h v = p[i];
        u32x2 d;
        d.x = pack4_fp8((float)v[0] * scale, (float)v[1] * scale,
                        (float)v[2] * scale, (float)v[3] * scale);
        d.y = pack4_fp8((float)v[4] * scale, (float)v[5] * scale,
                        (float)v[6] * scale, (float)v[7] * scale);
        q[i] = d;
    }
}

__global__ __launch_bounds__(256) void gemm_fp8_kernel(
    const unsigned char* __restrict__ A,
    const unsigned char* __restrict__ B,
    const _Float16* __restrict__ bias,
    _Float16* __restrict__ out,
    const unsigned int* __restrict__ amaxA,
    const unsigned int* __restrict__ amaxB,
    int M, int N, int K) {
    constexpr int LDSS = 144;
    constexpr int TILE_BYTES = 128 * LDSS;
    __shared__ unsigned char lds[2 * 2 * TILE_BYTES];

    const int t      = threadIdx.x;
    const int lane   = t & 31;
    const int wave   = t >> 5;
    const int waveM  = wave & 1;
    const int waveN  = wave >> 1;
    const int laneLo = lane & 15;
    const int laneHi = lane >> 4;

    const int blockM = blockIdx.y * 128;
    const int blockN = blockIdx.x * 128;

    const int ldRow = t >> 3;
    const int ldCol = (t & 7) * 16;

    const unsigned ldsBase = (unsigned)(uintptr_t)(void*)lds;

    auto issueTile = [&](int buf, int k0) {
        unsigned aBase = ldsBase + (unsigned)(buf * 2 * TILE_BYTES);
        unsigned bBase = aBase + (unsigned)TILE_BYTES;
#pragma unroll
        for (int p = 0; p < 4; ++p) {
            int r = p * 32 + ldRow;
            unsigned lofA = aBase + (unsigned)(r * LDSS + ldCol);
            unsigned gofA = (unsigned)(blockM + r) * (unsigned)K + (unsigned)(k0 + ldCol);
            asm volatile("global_load_async_to_lds_b128 %0, %1, %2"
                         :: "v"(lofA), "v"(gofA), "s"(A) : "memory");
            unsigned lofB = bBase + (unsigned)(r * LDSS + ldCol);
            unsigned gofB = (unsigned)(blockN + r) * (unsigned)K + (unsigned)(k0 + ldCol);
            asm volatile("global_load_async_to_lds_b128 %0, %1, %2"
                         :: "v"(lofB), "v"(gofB), "s"(B) : "memory");
        }
    };

    v8f acc[4][2] = {};

    const int nIter = K / 128;
    issueTile(0, 0);

    for (int it = 0; it < nIter; ++it) {
        const int buf = it & 1;
        asm volatile("s_wait_asynccnt 0x0" ::: "memory");
        __syncthreads();
        if (it + 1 < nIter) issueTile(1 - buf, (it + 1) * 128);

        const unsigned char* lA = lds + buf * 2 * TILE_BYTES;
        const unsigned char* lB = lA + TILE_BYTES;

        v16i bf[2];
#pragma unroll
        for (int nt = 0; nt < 2; ++nt) {
            const unsigned char* bb =
                lB + (waveN * 32 + nt * 16 + laneLo) * LDSS + laneHi * 16;
#pragma unroll
            for (int c = 0; c < 4; ++c) {
                u32x4 d = *(const u32x4*)(bb + c * 32);
                bf[nt][4 * c + 0] = (int)d.x;
                bf[nt][4 * c + 1] = (int)d.y;
                bf[nt][4 * c + 2] = (int)d.z;
                bf[nt][4 * c + 3] = (int)d.w;
            }
        }
        v16i af[4];
#pragma unroll
        for (int mt = 0; mt < 4; ++mt) {
            const unsigned char* ab =
                lA + (waveM * 64 + mt * 16 + laneLo) * LDSS + laneHi * 16;
#pragma unroll
            for (int j = 0; j < 8; ++j) {
                u32x2 d = *(const u32x2*)(ab + (j >> 1) * 32 + (j & 1) * 8);
                af[mt][2 * j]     = (int)d.x;
                af[mt][2 * j + 1] = (int)d.y;
            }
        }
        __builtin_amdgcn_sched_barrier(0);

#pragma unroll
        for (int mt = 0; mt < 4; ++mt)
#pragma unroll
            for (int nt = 0; nt < 2; ++nt)
                acc[mt][nt] = __builtin_amdgcn_wmma_f32_16x16x128_fp8_fp8(
                    af[mt], bf[nt], (short)0, acc[mt][nt], false, false);

        __syncthreads();
    }

    float aA = fmaxf(__uint_as_float(*amaxA), 1e-12f);
    float aB = fmaxf(__uint_as_float(*amaxB), 1e-12f);
    float sc = (aA / FP8_MAX) * (aB / FP8_MAX);
#pragma unroll
    for (int mt = 0; mt < 4; ++mt) {
#pragma unroll
        for (int nt = 0; nt < 2; ++nt) {
            int col  = blockN + waveN * 32 + nt * 16 + laneLo;
            float bv = (float)bias[col];
            int row0 = blockM + waveM * 64 + mt * 16 + laneHi * 8;
#pragma unroll
            for (int r = 0; r < 8; ++r) {
                float v = acc[mt][nt][r] * sc + bv;
                out[(size_t)(row0 + r) * N + col] = (_Float16)v;
            }
        }
    }
}

extern "C" void kernel_launch(void* const* d_in, const int* in_sizes, int n_in,
                              void* d_out, int out_size, void* d_ws, size_t ws_size,
                              hipStream_t stream) {
    const _Float16* x    = (const _Float16*)d_in[0];
    const _Float16* w    = (const _Float16*)d_in[1];
    const _Float16* bias = (const _Float16*)d_in[2];
    _Float16* out = (_Float16*)d_out;

    const int N = in_sizes[2];
    const int K = in_sizes[1] / N;
    const int M = in_sizes[0] / K;

    unsigned int*  amax = (unsigned int*)d_ws;
    unsigned char* x8   = (unsigned char*)d_ws + 256;
    unsigned char* w8   = x8 + (size_t)M * K;

    init_amax_kernel<<<1, 32, 0, stream>>>(amax);
    amax_kernel<<<1024, 256, 0, stream>>>(x, (size_t)M * K / 8, amax + 0);
    amax_kernel<<<1024, 256, 0, stream>>>(w, (size_t)N * K / 8, amax + 1);
    quant_kernel<<<2048, 256, 0, stream>>>(x, x8, (size_t)M * K / 8, amax + 0);
    quant_kernel<<<2048, 256, 0, stream>>>(w, w8, (size_t)N * K / 8, amax + 1);

    dim3 grid(N / 128, M / 128);
    gemm_fp8_kernel<<<grid, 256, 0, stream>>>(x8, w8, bias, out, amax + 0, amax + 1, M, N, K);
}
